// CrossModalAttention_22488448762091
// MI455X (gfx1250) — matrix operand order repaired and checked by reading, not compiled, not run on hardware
//
#include <hip/hip_runtime.h>

// CDNA5 / gfx1250, wave32. bf16 WMMA path (v_wmma_f32_16x16x32_bf16).
// Fixed problem size: B=8, T=1024, D=1024.

typedef __attribute__((ext_vector_type(16))) __bf16 v16bf;
typedef __attribute__((ext_vector_type(8)))  __bf16 v8bf;
typedef __attribute__((ext_vector_type(4)))  __bf16 v4bf;
typedef __attribute__((ext_vector_type(8)))  float  v8f;

#define DD 1024      // feature dim (also K dim and ld for all bf16 matrices)
#define TTOT 8192    // B*T rows for the big GEMMs

// ---------------------------------------------------------------------------
// f32 -> bf16 elementwise conversion (vectorized x4)
// ---------------------------------------------------------------------------
__global__ __launch_bounds__(256) void cvt_bf16_kernel(const float* __restrict__ in,
                                                       __bf16* __restrict__ out, int n4) {
  int i = blockIdx.x * blockDim.x + threadIdx.x;
  if (i < n4) {
    float4 x = ((const float4*)in)[i];
    v4bf y;
    y.x = (__bf16)x.x; y.y = (__bf16)x.y; y.z = (__bf16)x.z; y.w = (__bf16)x.w;
    ((v4bf*)out)[i] = y;
  }
}

// ---------------------------------------------------------------------------
// Universal WMMA GEMM: out[m,n] = sum_k A[m,k] * Bsrc[n,k]   (+ optional 2nd pair)
//   - A row-major [M,1024] bf16; Bsrc row-major [1024,1024] bf16 (row index = n)
//   - N fixed 1024, K fixed 1024, lda=ldb=1024
//   - block = 256 threads (8 waves); wave tile 64x64 (4x4 WMMA tiles)
//   - workgroup tile: M=256 (4 waves) x N=128 (2 waves)
//   - grid: x = 1024/128 = 8, y = M/256, z = batches (strides abatch/bbatch/ofbatch)
//   - epilogue: optional bias+relu; optional f32 out (ldF), bf16 out (ld=1024),
//     bf16 transposed out VT[b][n][t] with b=m>>10, t=m&1023.
// ---------------------------------------------------------------------------
__global__ __launch_bounds__(256) void gemm_wmma(
    const __bf16* __restrict__ A1, const __bf16* __restrict__ B1,
    const __bf16* __restrict__ A2, const __bf16* __restrict__ B2,
    long abatch, long bbatch,
    const float* __restrict__ bias, int relu,
    float* __restrict__ outF, long ldF, long ofbatch,
    __bf16* __restrict__ outB,
    __bf16* __restrict__ outBT)
{
  const int lane = threadIdx.x & 31;
  const int wave = threadIdx.x >> 5;
  const int hf   = lane >> 4;     // half-wave selector
  const int l15  = lane & 15;
  const int wm   = wave & 3;      // 4 waves along M
  const int wn   = wave >> 2;     // 2 waves along N
  const int mBase = blockIdx.y * 256 + wm * 64;
  const int nBase = blockIdx.x * 128 + wn * 64;
  const long bz = blockIdx.z;

  v8f acc[4][4];
  const v8f vzero = {0.f,0.f,0.f,0.f,0.f,0.f,0.f,0.f};
#pragma unroll
  for (int i = 0; i < 4; i++)
#pragma unroll
    for (int j = 0; j < 4; j++) acc[i][j] = vzero;

  const int npairs = (A2 != nullptr) ? 2 : 1;
  for (int p = 0; p < npairs; p++) {
    const __bf16* A = (p == 0 ? A1 : A2) + bz * abatch;
    const __bf16* B = (p == 0 ? B1 : B2) + bz * bbatch;
    // lane-resolved row base pointers
    const __bf16* aRow0 = A + (size_t)(mBase + l15) * DD;
    const __bf16* bRow0 = B + (size_t)(nBase + l15) * DD;

    for (int k0 = 0; k0 < DD; k0 += 32) {
      v16bf af[4], bfr[4];
      // A fragment 16x32: lane<16 -> K {0..7, 16..23}; lane>=16 -> K {8..15, 24..31}
#pragma unroll
      for (int i = 0; i < 4; i++) {
        const __bf16* ap = aRow0 + (size_t)i * 16 * DD + k0;
        v8bf lo = *(const v8bf*)(ap + 8 * hf);
        v8bf hi = *(const v8bf*)(ap + 16 + 8 * hf);
        af[i] = __builtin_shufflevector(lo, hi, 0,1,2,3,4,5,6,7,8,9,10,11,12,13,14,15);
      }
      // B fragment 32x16 (B[k,n] = Bsrc[n,k]): lane<16 -> K {0..7, 16..23}; lane>=16 -> K {8..15, 24..31}
#pragma unroll
      for (int j = 0; j < 4; j++) {
        const __bf16* bp = bRow0 + (size_t)j * 16 * DD + k0;
        v8bf lo = *(const v8bf*)(bp + 8 * hf);
        v8bf hi = *(const v8bf*)(bp + 16 + 8 * hf);
        bfr[j] = __builtin_shufflevector(lo, hi, 0,1,2,3,4,5,6,7,8,9,10,11,12,13,14,15);
      }
#pragma unroll
      for (int i = 0; i < 4; i++)
#pragma unroll
        for (int j = 0; j < 4; j++)
          acc[i][j] = __builtin_amdgcn_wmma_f32_16x16x32_bf16(
              false, af[i], false, bfr[j], (short)0, acc[i][j], false, false);
    }
  }

  // Epilogue. C/D layout: elem v -> row m = v + 8*hf, col n = l15.
#pragma unroll
  for (int i = 0; i < 4; i++) {
#pragma unroll
    for (int j = 0; j < 4; j++) {
      const int n = nBase + j * 16 + l15;
      const float bval = bias ? bias[n] : 0.f;
#pragma unroll
      for (int v = 0; v < 8; v++) {
        const int m = mBase + i * 16 + v + 8 * hf;
        float r = acc[i][j][v] + bval;
        if (relu) r = fmaxf(r, 0.f);
        if (outF)  outF[bz * ofbatch + (size_t)m * ldF + n] = r;
        if (outB)  outB[(size_t)m * DD + n] = (__bf16)r;
        if (outBT) outBT[((size_t)(m >> 10) << 20) + ((size_t)n << 10) + (size_t)(m & 1023)] = (__bf16)r;
      }
    }
  }
}

// ---------------------------------------------------------------------------
// Row softmax over 1024 f32 elements -> bf16 probabilities. One block per row.
// ---------------------------------------------------------------------------
__global__ __launch_bounds__(256) void softmax_kernel(const float* __restrict__ S,
                                                      __bf16* __restrict__ P) {
  __shared__ float red[256];
  const size_t row = blockIdx.x;
  const int t = threadIdx.x;
  float4 x = ((const float4*)(S + (row << 10)))[t];

  float m = fmaxf(fmaxf(x.x, x.y), fmaxf(x.z, x.w));
  red[t] = m; __syncthreads();
  for (int o = 128; o > 0; o >>= 1) {
    if (t < o) red[t] = fmaxf(red[t], red[t + o]);
    __syncthreads();
  }
  m = red[0]; __syncthreads();

  float e0 = __expf(x.x - m), e1 = __expf(x.y - m);
  float e2 = __expf(x.z - m), e3 = __expf(x.w - m);
  red[t] = e0 + e1 + e2 + e3; __syncthreads();
  for (int o = 128; o > 0; o >>= 1) {
    if (t < o) red[t] += red[t + o];
    __syncthreads();
  }
  const float inv = 1.f / red[0];

  v4bf y;
  y.x = (__bf16)(e0 * inv); y.y = (__bf16)(e1 * inv);
  y.z = (__bf16)(e2 * inv); y.w = (__bf16)(e3 * inv);
  ((v4bf*)(P + (row << 10)))[t] = y;
}

// ---------------------------------------------------------------------------
// Launch. Workspace layout (bytes), total 360 MB:
//   [0,48M)    f1..f3 bf16 (reused later for softmax P, also 48 MB)
//   [48,72M)   12 weights bf16 (2 MB each, d_in order)
//   [72,120M)  f1p..f3p bf16
//   [120,168M) q1..q3 bf16
//   [168,216M) k1..k3 bf16
//   [216,264M) vT1..vT3 bf16 (V stored transposed per batch: VT[b][d][t])
//   [264,360M) scores f32 [3][8][1024][1024]
// ---------------------------------------------------------------------------
extern "C" void kernel_launch(void* const* d_in, const int* in_sizes, int n_in,
                              void* d_out, int out_size, void* d_ws, size_t ws_size,
                              hipStream_t stream) {
  const float* f[3];  for (int i = 0; i < 3;  i++) f[i]  = (const float*)d_in[i];
  const float* W[12]; for (int j = 0; j < 12; j++) W[j]  = (const float*)d_in[3 + j];
  const float* bp[3]; for (int i = 0; i < 3;  i++) bp[i] = (const float*)d_in[15 + i];
  float* out = (float*)d_out;

  char* ws = (char*)d_ws;
  const size_t MB = 1u << 20;
  __bf16 *f_bf[3], *fp_bf[3], *q_bf[3], *k_bf[3], *vt_bf[3], *w_bf[12];
  for (int i = 0; i < 3;  i++) f_bf[i]  = (__bf16*)(ws + (size_t)i * 16 * MB);
  for (int j = 0; j < 12; j++) w_bf[j]  = (__bf16*)(ws + 48 * MB + (size_t)j * 2 * MB);
  for (int i = 0; i < 3;  i++) fp_bf[i] = (__bf16*)(ws + 72 * MB + (size_t)i * 16 * MB);
  for (int i = 0; i < 3;  i++) q_bf[i]  = (__bf16*)(ws + 120 * MB + (size_t)i * 16 * MB);
  for (int i = 0; i < 3;  i++) k_bf[i]  = (__bf16*)(ws + 168 * MB + (size_t)i * 16 * MB);
  for (int i = 0; i < 3;  i++) vt_bf[i] = (__bf16*)(ws + 216 * MB + (size_t)i * 16 * MB);
  float* scores = (float*)(ws + 264 * MB);
  __bf16* p_bf = (__bf16*)ws;  // reuse f_bf region after preproc phase

  const dim3 blk(256, 1, 1);

  // Phase 0: bf16 conversions
  for (int i = 0; i < 3; i++)
    cvt_bf16_kernel<<<dim3(8192, 1, 1), blk, 0, stream>>>(f[i], f_bf[i], (8 * 1024 * 1024) / 4);
  for (int j = 0; j < 12; j++)
    cvt_bf16_kernel<<<dim3(1024, 1, 1), blk, 0, stream>>>(W[j], w_bf[j], (1024 * 1024) / 4);

  // Phase 1: preproc  f_ip = relu(f_i @ Wp_i^T + bp_i)
  //   f32 -> d_out columns (3+i)*1024.., bf16 copy -> fp_bf
  const dim3 gBig(8, TTOT / 256, 1);
  for (int i = 0; i < 3; i++)
    gemm_wmma<<<gBig, blk, 0, stream>>>(f_bf[i], w_bf[i], nullptr, nullptr, 0L, 0L,
        bp[i], 1, out + (size_t)(3 + i) * 1024, 6144L, 0L, fp_bf[i], nullptr);

  // Phase 2: QKV projections (V stored transposed)
  for (int i = 0; i < 3; i++) {
    gemm_wmma<<<gBig, blk, 0, stream>>>(fp_bf[i], w_bf[3 + 3 * i], nullptr, nullptr, 0L, 0L,
        nullptr, 0, nullptr, 0L, 0L, q_bf[i], nullptr);
    gemm_wmma<<<gBig, blk, 0, stream>>>(fp_bf[i], w_bf[4 + 3 * i], nullptr, nullptr, 0L, 0L,
        nullptr, 0, nullptr, 0L, 0L, k_bf[i], nullptr);
    gemm_wmma<<<gBig, blk, 0, stream>>>(fp_bf[i], w_bf[5 + 3 * i], nullptr, nullptr, 0L, 0L,
        nullptr, 0, nullptr, 0L, 0L, nullptr, vt_bf[i]);
  }

  // Phase 3: scores (batched over B=8, dual A/B pairs summed)
  const dim3 gAtt(8, 4, 8);
  const long TD = 1L << 20;  // T*D elements per batch
  // s1 = q2 k1^T + q3 k1^T
  gemm_wmma<<<gAtt, blk, 0, stream>>>(q_bf[1], k_bf[0], q_bf[2], k_bf[0], TD, TD,
      nullptr, 0, scores + 0 * (8L << 20), 1024L, TD, nullptr, nullptr);
  // s2 = q1 k3^T + q3 k2^T
  gemm_wmma<<<gAtt, blk, 0, stream>>>(q_bf[0], k_bf[2], q_bf[2], k_bf[1], TD, TD,
      nullptr, 0, scores + 1 * (8L << 20), 1024L, TD, nullptr, nullptr);
  // s3 = q1 k3^T + q2 k3^T
  gemm_wmma<<<gAtt, blk, 0, stream>>>(q_bf[0], k_bf[2], q_bf[1], k_bf[2], TD, TD,
      nullptr, 0, scores + 2 * (8L << 20), 1024L, TD, nullptr, nullptr);

  // Phase 4: softmax over all 3*8*1024 rows -> bf16 P
  softmax_kernel<<<dim3(3 * 8 * 1024, 1, 1), blk, 0, stream>>>(scores, p_bf);

  // Phase 5: f_iu = P_i @ V_i  (B[k,n] = VT[n,k]) -> d_out columns i*1024..
  for (int i = 0; i < 3; i++)
    gemm_wmma<<<gAtt, blk, 0, stream>>>(p_bf + (size_t)i * (8L << 20), vt_bf[i],
        nullptr, nullptr, TD, TD,
        nullptr, 0, out + (size_t)i * 1024, 6144L, 1024L * 6144L, nullptr, nullptr);
}
